// MambaBaseline_44976897523939
// MI455X (gfx1250) — compile-verified, matrix operand order checked by reading, not run on hardware
//
#include <hip/hip_runtime.h>
#include <hip/hip_bf16.h>
#include <math.h>

// ---------------- model constants ----------------
#define BATCH   2
#define SEQ     1024
#define NTOK    (BATCH * SEQ)        // 2048 token rows, layout (s,b)
#define D_MODEL 768
#define D_INNER 1536
#define D_STATE 16
#define DT_RANK 48
#define D_CONV  4
#define VOCAB   32000
#define N_LAYERS 4
#define XPROJ_N (DT_RANK + 2 * D_STATE)   // 80

// ---------------- WMMA / TDM types ----------------
typedef __attribute__((ext_vector_type(16))) __bf16 v16bf;
typedef __attribute__((ext_vector_type(8)))  __bf16 bf16x8;
typedef __attribute__((ext_vector_type(8)))  float  v8f;
typedef unsigned int u32x4 __attribute__((ext_vector_type(4)));
typedef int          i32x8 __attribute__((ext_vector_type(8)));
typedef int          i32x4 __attribute__((ext_vector_type(4)));

// ---------------- GEMM tiling ----------------
#define BM 128
#define BN 128
#define BK 32
#define LDSROW 40   // bf16 elems per LDS row = 80 bytes (16B aligned, 20-bank stride)

// Build one wave-lane's v16bf fragment from an LDS tile row (ISA 16-bit A/B layout):
//  lanes 0-15  (hi=0): elems 0-7 = K 0-7,   elems 8-15 = K 16-23
//  lanes 16-31 (hi=1): elems 0-7 = K 8-15,  elems 8-15 = K 24-31
__device__ inline v16bf lds_frag(const __bf16* tile, int row, int hi) {
    const __bf16* p = tile + row * LDSROW;
    union { bf16x8 h[2]; v16bf v; } u;
    u.h[0] = *(const bf16x8*)(p + hi * 8);
    u.h[1] = *(const bf16x8*)(p + 16 + hi * 8);
    return u.v;
}

union B4 { __bf16 h[4]; uint2 u; };

// ---- A-side staging: 128x32 f32 chunk -> 4 float4 per thread ----
__device__ inline void load_tileA(const float* __restrict__ A, int lda, int K,
                                  int bm0, int k0, int tid, bool fullK, float4 a4[4])
{
    #pragma unroll
    for (int p = 0; p < 4; ++p) {
        int idx = tid + p * 256;          // 0..1023 float4 slots
        int row = idx >> 3;
        int c4  = (idx & 7) * 4;
        int gm  = bm0 + row;
        int gk  = k0 + c4;
        if (fullK) {
            a4[p] = *(const float4*)(A + (long)gm * lda + gk);
        } else {
            float ta[4];
            #pragma unroll
            for (int e = 0; e < 4; ++e) {
                int k  = gk + e;
                int kc = (k < K) ? k : (K - 1);
                float av = A[(long)gm * lda + kc];
                ta[e] = (k < K) ? av : 0.0f;
            }
            a4[p] = make_float4(ta[0], ta[1], ta[2], ta[3]);
        }
    }
}

__device__ inline void store_tileA(__bf16* sA, int tid, const float4 a4[4])
{
    #pragma unroll
    for (int p = 0; p < 4; ++p) {
        int idx = tid + p * 256;
        int row = idx >> 3;
        int c4  = (idx & 7) * 4;
        B4 ta;
        ta.h[0] = (__bf16)a4[p].x; ta.h[1] = (__bf16)a4[p].y;
        ta.h[2] = (__bf16)a4[p].z; ta.h[3] = (__bf16)a4[p].w;
        *(uint2*)(sA + row * LDSROW + c4) = ta.u;
    }
}

// ---- W-side staging (f32 fallback path) ----
__device__ inline void load_tileW(const float* __restrict__ W, int K, int N,
                                  int bn0, int k0, int tid, bool full, float4 w4[4])
{
    #pragma unroll
    for (int p = 0; p < 4; ++p) {
        int idx = tid + p * 256;
        int row = idx >> 3;
        int c4  = (idx & 7) * 4;
        int gn  = bn0 + row;
        int gk  = k0 + c4;
        if (full) {
            w4[p] = *(const float4*)(W + (long)gn * K + gk);
        } else {
            int gnc = (gn < N) ? gn : (N - 1);
            float tw[4];
            #pragma unroll
            for (int e = 0; e < 4; ++e) {
                int k  = gk + e;
                int kc = (k < K) ? k : (K - 1);
                float wv = W[(long)gnc * K + kc];
                tw[e] = (k < K && gn < N) ? wv : 0.0f;
            }
            w4[p] = make_float4(tw[0], tw[1], tw[2], tw[3]);
        }
    }
}

__device__ inline void store_tileW(__bf16* sB, int tid, const float4 w4[4])
{
    #pragma unroll
    for (int p = 0; p < 4; ++p) {
        int idx = tid + p * 256;
        int row = idx >> 3;
        int c4  = (idx & 7) * 4;
        B4 tw;
        tw.h[0] = (__bf16)w4[p].x; tw.h[1] = (__bf16)w4[p].y;
        tw.h[2] = (__bf16)w4[p].z; tw.h[3] = (__bf16)w4[p].w;
        *(uint2*)(sB + row * LDSROW + c4) = tw.u;
    }
}

// ---- TDM: DMA one bf16 weight tile (rows x 32, row stride K elems) into LDS ----
// D# per CDNA5 ISA. 2D tensor, data_size=2B,
// tile 32x128, zero-fill beyond (cols,rows), LDS pad 16B per 64B row -> 80B stride.
__device__ inline void tdm_load_b(const __bf16* gsrc, unsigned lds_off,
                                  unsigned rows, unsigned cols, unsigned strideK)
{
    unsigned long long ga = (unsigned long long)(uintptr_t)gsrc;
    u32x4 g0;
    g0[0] = 1u;                                            // count=1
    g0[1] = lds_off;                                       // lds_addr (bytes)
    g0[2] = (unsigned)ga;                                  // global_addr[31:0]
    g0[3] = (unsigned)((ga >> 32) & 0x01FFFFFFu) | (2u << 30);  // ga[56:32] | type=2
    i32x8 g1;
    g1[0] = (1 << 16) | (1 << 20) | (3 << 22) | (3 << 25); // data_size=2B, pad 16B/64B
    g1[1] = (int)((cols & 0xFFFFu) << 16);                 // tensor_dim0 lo
    g1[2] = (int)(((cols >> 16) & 0xFFFFu) | ((rows & 0xFFFFu) << 16));
    g1[3] = (int)(((rows >> 16) & 0xFFFFu) | (BK << 16));  // tile_dim0 = 32
    g1[4] = (int)BN;                                       // tile_dim1=128, tile_dim2=0
    g1[5] = (int)strideK;                                  // tensor_dim0_stride[31:0]
    g1[6] = 0;
    g1[7] = 0;
    i32x4 z4 = {0, 0, 0, 0};
    i32x8 z8 = {0, 0, 0, 0, 0, 0, 0, 0};
    // 6-arg form (upstream clang-23 / therock-10.0 headers)
    __builtin_amdgcn_tensor_load_to_lds(g0, g1, z4, z4, z8, 0);
}

// ---------------- epilogue helper ----------------
__device__ inline void store_c(float* __restrict__ C, const float* __restrict__ bias,
                               const float* __restrict__ R, int epi, int swz,
                               int M, int N, int bm0, int bn0, int wm, int wn,
                               int hi, int lh, const v8f acc[2][4])
{
    #pragma unroll
    for (int i = 0; i < 2; ++i) {
        #pragma unroll
        for (int j = 0; j < 4; ++j) {
            int nb = bn0 + wn + j * 16 + lh;
            if (nb >= N) continue;
            int mb = bm0 + wm + i * 16 + hi * 8;
            #pragma unroll
            for (int r = 0; r < 8; ++r) {
                int mrow = mb + r;
                if (mrow >= M) continue;
                float v = acc[i][j][r];
                if (epi == 1) {
                    v += bias[nb];
                    v = (v > 20.0f) ? v : log1pf(expf(v));   // softplus
                } else if (epi == 2) {
                    v += R[(long)mrow * N + nb];
                } else if (epi == 3) {
                    v += bias[nb];
                }
                long orow = mrow;
                if (swz) { int b = mrow & (BATCH - 1); int s = mrow >> 1; orow = (long)b * SEQ + s; }
                C[orow * (long)N + nb] = v;
            }
        }
    }
}

// ================= GEMM, bf16 weights + TDM B-tile =================
// C[M,N] = A[M,K] (f32, stride lda) x Wb[N,K]^T (bf16)
__global__ __launch_bounds__(256)
void wmma_gemm_wb(const float* __restrict__ A, const __bf16* __restrict__ Wb,
                  float* __restrict__ C, int M, int N, int K, int lda,
                  const float* __restrict__ bias, const float* __restrict__ R,
                  int epi, int swz)
{
    __shared__ alignas(16) __bf16 sA[2][BM * LDSROW];
    __shared__ alignas(16) __bf16 sB[2][BN * LDSROW];

    const int tid  = threadIdx.x;
    const int lane = tid & 31;
    const int wid  = tid >> 5;
    const int wm   = (wid & 3) * 32;
    const int wn   = (wid >> 2) * 64;
    const int hi   = lane >> 4;
    const int lh   = lane & 15;

    const int bm0 = blockIdx.y * BM;
    const int bn0 = blockIdx.x * BN;
    const int nch = (K + BK - 1) / BK;
    const unsigned rows = (unsigned)(N - bn0);

    v8f acc[2][4] = {};
    float4 a4[4];

    // prologue: TDM B chunk 0 + stage A chunk 0
    if (wid == 0)
        tdm_load_b(Wb + (long)bn0 * K, (unsigned)(uintptr_t)&sB[0][0],
                   rows, (unsigned)K, (unsigned)K);
    load_tileA(A, lda, K, bm0, 0, tid, BK <= K, a4);
    store_tileA(sA[0], tid, a4);
    if (wid == 0) __builtin_amdgcn_s_wait_tensorcnt(0);
    __syncthreads();

    for (int ch = 0; ch < nch; ++ch) {
        const int cur = ch & 1;
        const bool more = (ch + 1 < nch);
        if (more) {
            int k0 = (ch + 1) * BK;
            if (wid == 0)
                tdm_load_b(Wb + (long)bn0 * K + k0, (unsigned)(uintptr_t)&sB[cur ^ 1][0],
                           rows, (unsigned)(K - k0), (unsigned)K);
            load_tileA(A, lda, K, bm0, k0, tid, k0 + BK <= K, a4);
        }

        v16bf af[2], bfr[4];
        #pragma unroll
        for (int i = 0; i < 2; ++i) af[i]  = lds_frag(sA[cur], wm + i * 16 + lh, hi);
        #pragma unroll
        for (int j = 0; j < 4; ++j) bfr[j] = lds_frag(sB[cur], wn + j * 16 + lh, hi);

        #pragma unroll
        for (int i = 0; i < 2; ++i)
            #pragma unroll
            for (int j = 0; j < 4; ++j)
                acc[i][j] = __builtin_amdgcn_wmma_f32_16x16x32_bf16(
                    false, af[i], false, bfr[j], (short)0, acc[i][j], false, false);

        if (more) {
            store_tileA(sA[cur ^ 1], tid, a4);
            if (wid == 0) __builtin_amdgcn_s_wait_tensorcnt(0);
        }
        __syncthreads();
    }

    store_c(C, bias, R, epi, swz, M, N, bm0, bn0, wm, wn, hi, lh, acc);
}

// ================= GEMM, f32 weights (fallback if workspace too small) =================
__global__ __launch_bounds__(256)
void wmma_gemm(const float* __restrict__ A, const float* __restrict__ W,
               float* __restrict__ C, int M, int N, int K, int lda,
               const float* __restrict__ bias, const float* __restrict__ R,
               int epi, int swz)
{
    __shared__ alignas(16) __bf16 sA[2][BM * LDSROW];
    __shared__ alignas(16) __bf16 sB[2][BN * LDSROW];

    const int tid  = threadIdx.x;
    const int lane = tid & 31;
    const int wid  = tid >> 5;
    const int wm   = (wid & 3) * 32;
    const int wn   = (wid >> 2) * 64;
    const int hi   = lane >> 4;
    const int lh   = lane & 15;

    const int bm0 = blockIdx.y * BM;
    const int bn0 = blockIdx.x * BN;
    const bool fullN = (bn0 + BN <= N);
    const int nch = (K + BK - 1) / BK;

    v8f acc[2][4] = {};
    float4 a4[4], w4[4];

    load_tileA(A, lda, K, bm0, 0, tid, BK <= K, a4);
    load_tileW(W, K, N, bn0, 0, tid, fullN && (BK <= K), w4);
    store_tileA(sA[0], tid, a4);
    store_tileW(sB[0], tid, w4);
    __syncthreads();

    for (int ch = 0; ch < nch; ++ch) {
        const int cur = ch & 1;
        const bool more = (ch + 1 < nch);
        if (more) {
            int k0 = (ch + 1) * BK;
            load_tileA(A, lda, K, bm0, k0, tid, k0 + BK <= K, a4);
            load_tileW(W, K, N, bn0, k0, tid, fullN && (k0 + BK <= K), w4);
        }

        v16bf af[2], bfr[4];
        #pragma unroll
        for (int i = 0; i < 2; ++i) af[i]  = lds_frag(sA[cur], wm + i * 16 + lh, hi);
        #pragma unroll
        for (int j = 0; j < 4; ++j) bfr[j] = lds_frag(sB[cur], wn + j * 16 + lh, hi);

        #pragma unroll
        for (int i = 0; i < 2; ++i)
            #pragma unroll
            for (int j = 0; j < 4; ++j)
                acc[i][j] = __builtin_amdgcn_wmma_f32_16x16x32_bf16(
                    false, af[i], false, bfr[j], (short)0, acc[i][j], false, false);

        if (more) { store_tileA(sA[cur ^ 1], tid, a4); store_tileW(sB[cur ^ 1], tid, w4); }
        __syncthreads();
    }

    store_c(C, bias, R, epi, swz, M, N, bm0, bn0, wm, wn, hi, lh, acc);
}

// ---------------- weight convert f32 -> bf16 (once per launch) ----------------
__global__ void cvtw_k(const float* __restrict__ s, __bf16* __restrict__ d, long n4)
{
    long i = (long)blockIdx.x * blockDim.x + threadIdx.x;
    if (i >= n4) return;
    float4 v = ((const float4*)s)[i];
    B4 t;
    t.h[0] = (__bf16)v.x; t.h[1] = (__bf16)v.y;
    t.h[2] = (__bf16)v.z; t.h[3] = (__bf16)v.w;
    ((uint2*)d)[i] = t.u;
}

// ---------------- embedding gather ----------------
__global__ void embed_k(const int* __restrict__ tokens, const float* __restrict__ emb,
                        float* __restrict__ x)
{
    int idx = blockIdx.x * blockDim.x + threadIdx.x;
    if (idx >= NTOK * D_MODEL) return;
    int r = idx / D_MODEL, d = idx - r * D_MODEL;
    int b = r & 1, s = r >> 1;
    int tok = tokens[b * SEQ + s];
    x[idx] = emb[(long)tok * D_MODEL + d];
}

// ---------------- layernorm: one wave32 per 768-wide row ----------------
__global__ __launch_bounds__(256)
void ln_k(const float* __restrict__ x, const float* __restrict__ g,
          const float* __restrict__ b, float* __restrict__ o)
{
    int row  = blockIdx.x * 8 + (threadIdx.x >> 5);
    int lane = threadIdx.x & 31;
    const float* xr = x + (long)row * D_MODEL;
    float s = 0.f, s2 = 0.f;
    for (int i = lane; i < D_MODEL; i += 32) { float v = xr[i]; s += v; s2 += v * v; }
    #pragma unroll
    for (int m = 16; m; m >>= 1) { s += __shfl_xor(s, m, 32); s2 += __shfl_xor(s2, m, 32); }
    float mean = s * (1.0f / D_MODEL);
    float var  = s2 * (1.0f / D_MODEL) - mean * mean;
    float rs   = rsqrtf(var + 1e-5f);
    float* orow = o + (long)row * D_MODEL;
    for (int i = lane; i < D_MODEL; i += 32)
        orow[i] = (xr[i] - mean) * rs * g[i] + b[i];
}

// ---------------- depthwise causal conv (width 4) + bias + SiLU ----------------
__global__ void conv_k(const float* __restrict__ xz, const float* __restrict__ w,
                       const float* __restrict__ cb, float* __restrict__ o)
{
    int idx = blockIdx.x * blockDim.x + threadIdx.x;
    if (idx >= NTOK * D_INNER) return;
    int r = idx / D_INNER, c = idx - r * D_INNER;
    int b = r & 1, s = r >> 1;
    float acc = cb[c];
    #pragma unroll
    for (int j = 0; j < D_CONV; ++j) {
        int sp = s - (D_CONV - 1) + j;
        if (sp >= 0)
            acc += w[c * D_CONV + j] * xz[(long)(sp * BATCH + b) * (2 * D_INNER) + c];
    }
    o[idx] = acc / (1.0f + expf(-acc));    // SiLU
}

// ---------------- selective scan: one thread per (b, channel), state in regs ----------------
__global__ __launch_bounds__(256)
void scan_k(const float* __restrict__ dt, const float* __restrict__ xdb,
            const float* __restrict__ xc, const float* __restrict__ A_log,
            float* __restrict__ y)
{
    __shared__ float sBC[2][2 * D_STATE];
    int g = blockIdx.x * blockDim.x + threadIdx.x;   // 0..3071
    int b = g / D_INNER;
    int c = g - b * D_INNER;

    float An[D_STATE], h[D_STATE];
    #pragma unroll
    for (int n = 0; n < D_STATE; ++n) { An[n] = -expf(A_log[c * D_STATE + n]); h[n] = 0.f; }

    for (int s = 0; s < SEQ; ++s) {
        int row = s * BATCH + b;
        if (threadIdx.x < 2 * D_STATE)
            sBC[s & 1][threadIdx.x] = xdb[(long)row * XPROJ_N + DT_RANK + threadIdx.x];
        __syncthreads();
        const float* Bs = sBC[s & 1];
        const float* Cs = Bs + D_STATE;
        float dtv = dt[(long)row * D_INNER + c];
        float dx  = dtv * xc[(long)row * D_INNER + c];
        float acc = 0.f;
        #pragma unroll
        for (int n = 0; n < D_STATE; ++n) {
            h[n] = h[n] * expf(dtv * An[n]) + dx * Bs[n];
            acc += h[n] * Cs[n];
        }
        y[(long)row * D_INNER + c] = acc;
    }
}

// ---------------- gate: y = (y + D*xc) * silu(z) ----------------
__global__ void gate_k(float* __restrict__ y, const float* __restrict__ xc,
                       const float* __restrict__ xz, const float* __restrict__ D)
{
    int idx = blockIdx.x * blockDim.x + threadIdx.x;
    if (idx >= NTOK * D_INNER) return;
    int r = idx / D_INNER, c = idx - r * D_INNER;
    float z = xz[(long)r * (2 * D_INNER) + D_INNER + c];
    y[idx] = (y[idx] + D[c] * xc[idx]) * (z / (1.0f + expf(-z)));
}

// ---------------- launcher ----------------
extern "C" void kernel_launch(void* const* d_in, const int* in_sizes, int n_in,
                              void* d_out, int out_size, void* d_ws, size_t ws_size,
                              hipStream_t stream)
{
    const int*   tokens    = (const int*)  d_in[0];
    const float* emb       = (const float*)d_in[1];
    const float* head_bias = (const float*)d_in[2];
    const float* ln_g      = (const float*)d_in[3];
    const float* ln_b      = (const float*)d_in[4];
    const float* W_in      = (const float*)d_in[5];
    const float* conv_w    = (const float*)d_in[6];
    const float* conv_b    = (const float*)d_in[7];
    const float* W_xproj   = (const float*)d_in[8];
    const float* W_dt      = (const float*)d_in[9];
    const float* b_dt      = (const float*)d_in[10];
    const float* A_log     = (const float*)d_in[11];
    const float* D_ssm     = (const float*)d_in[12];
    const float* W_out     = (const float*)d_in[13];
    const float* norm_g    = (const float*)d_in[14];
    const float* norm_b    = (const float*)d_in[15];
    float* out = (float*)d_out;

    // f32 activation workspace (~76 MB)
    const long N_X   = (long)NTOK * D_MODEL;
    const long N_XZ  = (long)NTOK * 2 * D_INNER;
    const long N_XC  = (long)NTOK * D_INNER;
    const long N_XDB = (long)NTOK * XPROJ_N;
    float* x     = (float*)d_ws;
    float* xln   = x     + N_X;
    float* xz    = xln   + N_X;
    float* xconv = xz    + N_XZ;
    float* xdb   = xconv + N_XC;
    float* dtb   = xdb   + N_XDB;
    float* yb    = dtb   + N_XC;
    float* f32_end = yb + N_XC;

    // bf16 weight cache (~79 MB), used if workspace allows
    const long NW_EMB = (long)VOCAB * D_MODEL;
    const long NW_IN  = (long)N_LAYERS * 2 * D_INNER * D_MODEL;
    const long NW_XP  = (long)N_LAYERS * XPROJ_N * D_INNER;
    const long NW_DT  = (long)N_LAYERS * D_INNER * DT_RANK;
    const long NW_OUT = (long)N_LAYERS * D_MODEL * D_INNER;
    const long NW_ALL = NW_EMB + NW_IN + NW_XP + NW_DT + NW_OUT;

    size_t f32_bytes = (size_t)((char*)f32_end - (char*)d_ws);
    bool use_bf = (ws_size >= f32_bytes + (size_t)NW_ALL * 2);

    __bf16* wb_emb = (__bf16*)f32_end;
    __bf16* wb_in  = wb_emb + NW_EMB;
    __bf16* wb_xp  = wb_in  + NW_IN;
    __bf16* wb_dt  = wb_xp  + NW_XP;
    __bf16* wb_out = wb_dt  + NW_DT;

    if (use_bf) {
        cvtw_k<<<(unsigned)((NW_EMB / 4 + 255) / 256), 256, 0, stream>>>(emb,     wb_emb, NW_EMB / 4);
        cvtw_k<<<(unsigned)((NW_IN  / 4 + 255) / 256), 256, 0, stream>>>(W_in,    wb_in,  NW_IN  / 4);
        cvtw_k<<<(unsigned)((NW_XP  / 4 + 255) / 256), 256, 0, stream>>>(W_xproj, wb_xp,  NW_XP  / 4);
        cvtw_k<<<(unsigned)((NW_DT  / 4 + 255) / 256), 256, 0, stream>>>(W_dt,    wb_dt,  NW_DT  / 4);
        cvtw_k<<<(unsigned)((NW_OUT / 4 + 255) / 256), 256, 0, stream>>>(W_out,   wb_out, NW_OUT / 4);
    }

    embed_k<<<(NTOK * D_MODEL + 255) / 256, 256, 0, stream>>>(tokens, emb, x);

    dim3 g1((2 * D_INNER) / BN, NTOK / BM);
    dim3 g2((XPROJ_N + BN - 1) / BN, NTOK / BM);
    dim3 g3(D_INNER / BN, NTOK / BM);
    dim3 g4(D_MODEL / BN, NTOK / BM);
    dim3 gh(VOCAB / BN, NTOK / BM);

    for (int i = 0; i < N_LAYERS; ++i) {
        ln_k<<<NTOK / 8, 256, 0, stream>>>(x, ln_g + i * D_MODEL, ln_b + i * D_MODEL, xln);

        if (use_bf)
            wmma_gemm_wb<<<g1, 256, 0, stream>>>(xln, wb_in + (long)i * 2 * D_INNER * D_MODEL, xz,
                NTOK, 2 * D_INNER, D_MODEL, D_MODEL, nullptr, nullptr, 0, 0);
        else
            wmma_gemm<<<g1, 256, 0, stream>>>(xln, W_in + (long)i * 2 * D_INNER * D_MODEL, xz,
                NTOK, 2 * D_INNER, D_MODEL, D_MODEL, nullptr, nullptr, 0, 0);

        conv_k<<<(NTOK * D_INNER + 255) / 256, 256, 0, stream>>>(
            xz, conv_w + (long)i * D_INNER * D_CONV, conv_b + i * D_INNER, xconv);

        if (use_bf)
            wmma_gemm_wb<<<g2, 256, 0, stream>>>(xconv, wb_xp + (long)i * XPROJ_N * D_INNER, xdb,
                NTOK, XPROJ_N, D_INNER, D_INNER, nullptr, nullptr, 0, 0);
        else
            wmma_gemm<<<g2, 256, 0, stream>>>(xconv, W_xproj + (long)i * XPROJ_N * D_INNER, xdb,
                NTOK, XPROJ_N, D_INNER, D_INNER, nullptr, nullptr, 0, 0);

        if (use_bf)
            wmma_gemm_wb<<<g3, 256, 0, stream>>>(xdb, wb_dt + (long)i * D_INNER * DT_RANK, dtb,
                NTOK, D_INNER, DT_RANK, XPROJ_N, b_dt + i * D_INNER, nullptr, 1, 0);
        else
            wmma_gemm<<<g3, 256, 0, stream>>>(xdb, W_dt + (long)i * D_INNER * DT_RANK, dtb,
                NTOK, D_INNER, DT_RANK, XPROJ_N, b_dt + i * D_INNER, nullptr, 1, 0);

        scan_k<<<(BATCH * D_INNER) / 256, 256, 0, stream>>>(
            dtb, xdb, xconv, A_log + (long)i * D_INNER * D_STATE, yb);

        gate_k<<<(NTOK * D_INNER + 255) / 256, 256, 0, stream>>>(
            yb, xconv, xz, D_ssm + i * D_INNER);

        if (use_bf)
            wmma_gemm_wb<<<g4, 256, 0, stream>>>(yb, wb_out + (long)i * D_MODEL * D_INNER, x,
                NTOK, D_MODEL, D_INNER, D_INNER, nullptr, x, 2, 0);
        else
            wmma_gemm<<<g4, 256, 0, stream>>>(yb, W_out + (long)i * D_MODEL * D_INNER, x,
                NTOK, D_MODEL, D_INNER, D_INNER, nullptr, x, 2, 0);
    }

    ln_k<<<NTOK / 8, 256, 0, stream>>>(x, norm_g, norm_b, xln);

    if (use_bf)
        wmma_gemm_wb<<<gh, 256, 0, stream>>>(xln, wb_emb, out, NTOK, VOCAB, D_MODEL, D_MODEL,
                                             head_bias, nullptr, 3, 1);
    else
        wmma_gemm<<<gh, 256, 0, stream>>>(xln, emb, out, NTOK, VOCAB, D_MODEL, D_MODEL,
                                          head_bias, nullptr, 3, 1);
}
